// EncoderLayer_64321430225657
// MI455X (gfx1250) — compile-verified, matrix operand order checked by reading, not run on hardware
//
#include <hip/hip_runtime.h>
#include <hip/hip_bf16.h>
#include <math.h>

#define DEV __device__ __forceinline__

typedef __attribute__((ext_vector_type(16))) __bf16 v16bf;
typedef __attribute__((ext_vector_type(8)))  __bf16 v8bf;
typedef __attribute__((ext_vector_type(8)))  float  v8f;
typedef __attribute__((ext_vector_type(4)))  int    v4i;

constexpr int Bb   = 8;
constexpr int Ls   = 1024;
constexpr int Dd   = 512;
constexpr int Hh   = 8;
constexpr int DKk  = 64;
constexpr int DVv  = 64;
constexpr int DFFf = 2048;
constexpr int Uu   = 35;      // min(5*ceil(ln(1024)), 1024) = 35
constexpr int ML   = Bb * Ls; // 8192 rows in all big GEMMs

DEV __bf16 f2bf(float f) {
  unsigned u = __builtin_bit_cast(unsigned, f);
  u += 0x7FFFu + ((u >> 16) & 1u);          // round-to-nearest-even
  unsigned short hs = (unsigned short)(u >> 16);
  return __builtin_bit_cast(__bf16, hs);
}

DEV float gelu_exact(float x) { return 0.5f * x * (1.0f + erff(x * 0.7071067811865475f)); }

// ---------------------------------------------------------------------------
// bf16 GEMM: C[M,N] = act(A[M,K] @ B^T + bias) (+ residual)
// A: bf16 M x K row-major.  B: bf16 N x K row-major (K contiguous).
// One wave computes a 32x32 C tile = 2x2 WMMA tiles; 8 waves -> 64x128 block.
// Fragment = two global_load_b128 of 8 contiguous bf16 each, matching the
// CDNA5 16-bit fragment layout (ISA 7.12.2): K = 8*half..+7 then 16+8*half..+7.
// OUT bit0: write f32 C.  OUT bit1: write bf16 C.
// ---------------------------------------------------------------------------
template<int ACT, int OUT, int RES>
__global__ __launch_bounds__(256)
void gemm_bf16_kernel(const __bf16* __restrict__ A, const __bf16* __restrict__ Bm,
                      const float* __restrict__ bias, const float* __restrict__ res,
                      float* __restrict__ Cf, __bf16* __restrict__ Cb,
                      int M, int N, int K)
{
  const int lane = threadIdx.x;      // wave32
  const int wave = threadIdx.y;      // 0..7
  const int half = lane >> 4;
  const int r    = lane & 15;
  const int wy   = wave >> 2;        // 0..1
  const int wx   = wave & 3;         // 0..3
  const int m0   = blockIdx.y * 64  + wy * 32;
  const int n0   = blockIdx.x * 128 + wx * 32;
  if (m0 >= M || n0 >= N) return;    // wave-uniform

  v8f acc[2][2] = {{{}, {}}, {{}, {}}};
  const __bf16* Ar0 = A  + (size_t)(m0 + r) * K;
  const __bf16* Ar1 = Ar0 + (size_t)16 * K;
  const __bf16* Br0 = Bm + (size_t)(n0 + r) * K;
  const __bf16* Br1 = Br0 + (size_t)16 * K;
  const int koff = half * 8;

  for (int k0 = 0; k0 < K; k0 += 32) {
    __builtin_prefetch(Ar0 + k0 + 512, 0, 1);   // global_prefetch_b8
    v8bf a0l = *(const v8bf*)(Ar0 + k0 + koff);
    v8bf a0h = *(const v8bf*)(Ar0 + k0 + 16 + koff);
    v8bf a1l = *(const v8bf*)(Ar1 + k0 + koff);
    v8bf a1h = *(const v8bf*)(Ar1 + k0 + 16 + koff);
    v8bf b0l = *(const v8bf*)(Br0 + k0 + koff);
    v8bf b0h = *(const v8bf*)(Br0 + k0 + 16 + koff);
    v8bf b1l = *(const v8bf*)(Br1 + k0 + koff);
    v8bf b1h = *(const v8bf*)(Br1 + k0 + 16 + koff);
    v16bf a0 = __builtin_shufflevector(a0l, a0h, 0,1,2,3,4,5,6,7,8,9,10,11,12,13,14,15);
    v16bf a1 = __builtin_shufflevector(a1l, a1h, 0,1,2,3,4,5,6,7,8,9,10,11,12,13,14,15);
    v16bf b0 = __builtin_shufflevector(b0l, b0h, 0,1,2,3,4,5,6,7,8,9,10,11,12,13,14,15);
    v16bf b1 = __builtin_shufflevector(b1l, b1h, 0,1,2,3,4,5,6,7,8,9,10,11,12,13,14,15);
    acc[0][0] = __builtin_amdgcn_wmma_f32_16x16x32_bf16(false, a0, false, b0, (short)0, acc[0][0], false, false);
    acc[0][1] = __builtin_amdgcn_wmma_f32_16x16x32_bf16(false, a0, false, b1, (short)0, acc[0][1], false, false);
    acc[1][0] = __builtin_amdgcn_wmma_f32_16x16x32_bf16(false, a1, false, b0, (short)0, acc[1][0], false, false);
    acc[1][1] = __builtin_amdgcn_wmma_f32_16x16x32_bf16(false, a1, false, b1, (short)0, acc[1][1], false, false);
  }

#pragma unroll
  for (int i = 0; i < 2; ++i) {
#pragma unroll
    for (int j = 0; j < 2; ++j) {
      const int col = n0 + j * 16 + r;
      const float bvv = bias[col];
#pragma unroll
      for (int g = 0; g < 8; ++g) {           // C VGPR g: M = g (+8 upper lanes)
        const int row = m0 + i * 16 + g + half * 8;
        float vv = acc[i][j][g] + bvv;
        if (ACT == 1) vv = gelu_exact(vv);
        if (RES)      vv += res[(size_t)row * N + col];
        if (OUT & 1)  Cf[(size_t)row * N + col] = vv;
        if (OUT & 2)  Cb[(size_t)row * N + col] = f2bf(vv);
      }
    }
  }
}

// --------------------- fp32 -> bf16 packers --------------------------------
__global__ __launch_bounds__(256)
void pack_convert_kernel(const float* __restrict__ in, __bf16* __restrict__ out, int n)
{
  int t = blockIdx.x * 256 + threadIdx.x;
  if (t < n) out[t] = f2bf(in[t]);
}

// in: K x N f32 row-major   ->  out: N x K bf16 row-major (transpose)
__global__ __launch_bounds__(256)
void pack_transpose_kernel(const float* __restrict__ in, __bf16* __restrict__ out,
                           int K, int N)
{
  int t = blockIdx.x * 256 + threadIdx.x;
  if (t >= K * N) return;
  int k = t % K;
  int n = t / K;
  out[t] = f2bf(in[(size_t)k * N + n]);
}

// --------------------- sampled QK scores -> M metric -----------------------
__global__ __launch_bounds__(256)
void sample_score_kernel(const float* __restrict__ q, const float* __restrict__ k,
                         const int* __restrict__ idx, float* __restrict__ Msc)
{
  int t = blockIdx.x * 256 + threadIdx.x;      // (b,h,l), l fastest
  if (t >= Bb * Hh * Ls) return;
  int l = t % Ls;
  int h = (t / Ls) % Hh;
  int b = t / (Ls * Hh);
  const float* qr = q + (size_t)(b * Ls + l) * (Hh * DKk) + h * DKk;
  float qv[DKk];
#pragma unroll
  for (int d = 0; d < DKk; ++d) qv[d] = qr[d];
  float mx = -3.4e38f, sm = 0.0f;
  for (int u = 0; u < Uu; ++u) {
    int j = idx[l * Uu + u];
    const float* kr = k + (size_t)(b * Ls + j) * (Hh * DKk) + h * DKk;
    float s = 0.0f;
#pragma unroll
    for (int d = 0; d < DKk; ++d) s += qv[d] * kr[d];
    mx = fmaxf(mx, s);
    sm += s;
  }
  Msc[t] = mx - sm * (1.0f / Uu);
}

// --------------------- iterative top-U argmax per (b,h) --------------------
__global__ __launch_bounds__(256)
void topk_kernel(const float* __restrict__ Msc, int* __restrict__ mtop)
{
  __shared__ float sv[Ls];
  __shared__ float rv[256];
  __shared__ int   ri[256];
  const int bh = blockIdx.x;
  const int t  = threadIdx.x;
#if __has_builtin(__builtin_amdgcn_global_load_async_to_lds_b128) && \
    __has_builtin(__builtin_amdgcn_s_wait_asynccnt)
  {
    // CDNA5 async global->LDS copy (ASYNCcnt-tracked): 256 lanes x 16B = 4KB row
    __builtin_amdgcn_global_load_async_to_lds_b128(
        (__attribute__((address_space(1))) v4i*)(Msc + (size_t)bh * Ls + t * 4),
        (__attribute__((address_space(3))) v4i*)(&sv[t * 4]),
        0, 0);
    __builtin_amdgcn_s_wait_asynccnt(0);
  }
#else
  for (int i = t; i < Ls; i += 256) sv[i] = Msc[(size_t)bh * Ls + i];
#endif
  __syncthreads();
  for (int s = 0; s < Uu; ++s) {
    float best = -3.4e38f; int bi = Ls;
    for (int i = t; i < Ls; i += 256) {
      float v = sv[i];
      if (v > best) { best = v; bi = i; }      // ascending scan keeps lowest idx on ties
    }
    rv[t] = best; ri[t] = bi;
    __syncthreads();
    for (int w = 128; w > 0; w >>= 1) {
      if (t < w) {
        float ov = rv[t + w]; int oi = ri[t + w];
        if (ov > rv[t] || (ov == rv[t] && oi < ri[t])) { rv[t] = ov; ri[t] = oi; }
      }
      __syncthreads();
    }
    if (t == 0) { mtop[bh * Uu + s] = ri[0]; sv[ri[0]] = -3.4e38f; }
    __syncthreads();
  }
}

// --------------------- mean of v over L per (b,h,d) ------------------------
__global__ __launch_bounds__(256)
void vmean_kernel(const float* __restrict__ v, float* __restrict__ vmean)
{
  int t = blockIdx.x * 256 + threadIdx.x;
  if (t >= Bb * Hh * DVv) return;
  int d = t % DVv;
  int h = (t / DVv) % Hh;
  int b = t / (DVv * Hh);
  const float* base = v + (size_t)b * Ls * (Hh * DVv) + h * DVv + d;
  float s = 0.0f;
  for (int l = 0; l < Ls; ++l) s += base[(size_t)l * (Hh * DVv)];
  vmean[t] = s * (1.0f / Ls);
}

// ---------- full attention for the U selected queries of each (b,h) --------
__global__ __launch_bounds__(256)
void attn_kernel(const float* __restrict__ q, const float* __restrict__ k,
                 const float* __restrict__ v, const int* __restrict__ mtop,
                 float* __restrict__ upd)
{
  __shared__ float qs[DKk];
  __shared__ float ps[Ls];
  __shared__ float red[256];
  __shared__ float part[4][DVv];
  const int blk = blockIdx.x;                  // (b,h,u)
  const int u = blk % Uu;
  const int h = (blk / Uu) % Hh;
  const int b = blk / (Uu * Hh);
  const int t = threadIdx.x;
  const int qrow = mtop[(b * Hh + h) * Uu + u];
  if (t < DKk) qs[t] = q[(size_t)(b * Ls + qrow) * (Hh * DKk) + h * DKk + t];
  __syncthreads();
  float lmax = -3.4e38f;
  for (int l = t; l < Ls; l += 256) {
    const float* kr = k + (size_t)(b * Ls + l) * (Hh * DKk) + h * DKk;
    float s = 0.0f;
#pragma unroll
    for (int d = 0; d < DKk; ++d) s += qs[d] * kr[d];
    s *= 0.125f;                               // 1/sqrt(64)
    ps[l] = s;
    lmax = fmaxf(lmax, s);
  }
  red[t] = lmax;
  __syncthreads();
  for (int w = 128; w > 0; w >>= 1) {
    if (t < w) red[t] = fmaxf(red[t], red[t + w]);
    __syncthreads();
  }
  const float mx = red[0];
  __syncthreads();
  float lsum = 0.0f;
  for (int l = t; l < Ls; l += 256) {
    float e = __expf(ps[l] - mx);
    ps[l] = e;
    lsum += e;
  }
  red[t] = lsum;
  __syncthreads();
  for (int w = 128; w > 0; w >>= 1) {
    if (t < w) red[t] += red[t + w];
    __syncthreads();
  }
  const float inv = 1.0f / red[0];
  const int d = t & 63;
  const int c = t >> 6;                        // 4 chunks of 256 l's
  float acc = 0.0f;
  for (int l = c * 256; l < (c + 1) * 256; ++l)
    acc += ps[l] * v[(size_t)(b * Ls + l) * (Hh * DVv) + h * DVv + d];
  part[c][d] = acc;
  __syncthreads();
  if (t < DVv) {
    float s = (part[0][t] + part[1][t] + part[2][t] + part[3][t]) * inv;
    upd[((size_t)(b * Hh + h) * Uu + u) * DVv + t] = s;
  }
}

// ------- ctx (bf16, direct WMMA input) = broadcast(v-mean) + scatter -------
__global__ __launch_bounds__(256)
void ctx_bcast_kernel(const float* __restrict__ vmean, __bf16* __restrict__ ctx)
{
  size_t t = (size_t)blockIdx.x * 256 + threadIdx.x;
  if (t >= (size_t)Bb * Ls * Hh * DVv) return;
  int d = (int)(t % DVv);
  int h = (int)((t / DVv) % Hh);
  int b = (int)(t / ((size_t)DVv * Hh * Ls));
  ctx[t] = f2bf(vmean[(b * Hh + h) * DVv + d]);
}

__global__ __launch_bounds__(256)
void ctx_scatter_kernel(const int* __restrict__ mtop, const float* __restrict__ upd,
                        __bf16* __restrict__ ctx)
{
  int t = blockIdx.x * 256 + threadIdx.x;
  if (t >= Bb * Hh * Uu * DVv) return;
  int d = t % DVv;
  int u = (t / DVv) % Uu;
  int h = (t / (DVv * Uu)) % Hh;
  int b = t / (DVv * Uu * Hh);
  int l = mtop[(b * Hh + h) * Uu + u];
  ctx[(size_t)(b * Ls + l) * (Hh * DVv) + h * DVv + d] = f2bf(upd[t]);
}

// --------------------- LayerNorm over last dim (512) -----------------------
__global__ __launch_bounds__(256)
void layernorm_kernel(const float* __restrict__ z, const float* __restrict__ gamma,
                      const float* __restrict__ beta, float* __restrict__ out)
{
  __shared__ float r1[256];
  __shared__ float r2[256];
  const int row = blockIdx.x;
  const int t = threadIdx.x;
  const float a  = z[(size_t)row * Dd + t];
  const float b2 = z[(size_t)row * Dd + t + 256];
  r1[t] = a + b2;
  r2[t] = a * a + b2 * b2;
  __syncthreads();
  for (int w = 128; w > 0; w >>= 1) {
    if (t < w) { r1[t] += r1[t + w]; r2[t] += r2[t + w]; }
    __syncthreads();
  }
  const float mu  = r1[0] * (1.0f / Dd);
  const float var = r2[0] * (1.0f / Dd) - mu * mu;
  const float rs  = rsqrtf(var + 1e-5f);
  out[(size_t)row * Dd + t]       = (a  - mu) * rs * gamma[t]       + beta[t];
  out[(size_t)row * Dd + t + 256] = (b2 - mu) * rs * gamma[t + 256] + beta[t + 256];
}

// --------------------- write m_top (twice) as float tail -------------------
__global__ __launch_bounds__(256)
void write_mtop_kernel(const int* __restrict__ mtop, float* __restrict__ outTail)
{
  int t = blockIdx.x * 256 + threadIdx.x;
  const int n = Bb * Hh * Uu;
  if (t < n) {
    float fv = (float)mtop[t];
    outTail[t]     = fv;
    outTail[n + t] = fv;
  }
}

// ---------------------------------------------------------------------------
extern "C" void kernel_launch(void* const* d_in, const int* in_sizes, int n_in,
                              void* d_out, int out_size, void* d_ws, size_t ws_size,
                              hipStream_t stream)
{
  const float* x     = (const float*)d_in[0];
  const float* Wq    = (const float*)d_in[1];
  const float* bq    = (const float*)d_in[2];
  const float* Wk    = (const float*)d_in[3];
  const float* bk    = (const float*)d_in[4];
  const float* Wv    = (const float*)d_in[5];
  const float* bvp   = (const float*)d_in[6];
  const float* Wo    = (const float*)d_in[7];
  const float* bo    = (const float*)d_in[8];
  const float* W1    = (const float*)d_in[9];   // (DFF, D): already N x K
  const float* b1    = (const float*)d_in[10];
  const float* W2    = (const float*)d_in[11];  // (D, DFF): already N x K
  const float* b2    = (const float*)d_in[12];
  const float* gamma = (const float*)d_in[13];
  const float* beta  = (const float*)d_in[14];
  const int*   idx   = (const int*)d_in[15];
  float* out = (float*)d_out;

  // ---- workspace layout ----
  const size_t SZ = (size_t)ML * Dd;            // 4,194,304
  float* ws    = (float*)d_ws;
  float* q     = ws;                            // f32 region
  float* kbuf  = q    + SZ;
  float* vbuf  = kbuf + SZ;
  float* attn  = vbuf + SZ;
  float* zbuf  = attn + SZ;
  float* Msc   = zbuf + SZ;
  float* vmean = Msc  + (size_t)Bb * Hh * Ls;
  float* upd   = vmean + Bb * Hh * DVv;
  int*   mtop  = (int*)(upd + (size_t)Bb * Hh * Uu * DVv);
  __bf16* bfbase = (__bf16*)(mtop + Bb * Hh * Uu + 8);  // keep 16B alignment
  __bf16* xbf    = bfbase;                      // ML*D
  __bf16* attnbf = xbf    + SZ;
  __bf16* ctxbf  = attnbf + SZ;
  __bf16* hbf    = ctxbf  + SZ;                 // ML*DFF
  __bf16* Wqp    = hbf  + (size_t)ML * DFFf;    // D*D each, transposed
  __bf16* Wkp    = Wqp  + (size_t)Dd * Dd;
  __bf16* Wvp    = Wkp  + (size_t)Dd * Dd;
  __bf16* Wop    = Wvp  + (size_t)Dd * Dd;
  __bf16* W1bf   = Wop  + (size_t)Dd * Dd;      // DFF*D
  __bf16* W2bf   = W1bf + (size_t)DFFf * Dd;    // D*DFF

  dim3 blk(32, 8);                              // 8 waves: 2x4 -> 64x128 C tile

  // ---- pack inputs / weights to bf16 (stateless, every call) ----
  pack_convert_kernel<<<(unsigned)(SZ / 256), 256, 0, stream>>>(x, xbf, (int)SZ);
  pack_transpose_kernel<<<(Dd * Dd) / 256, 256, 0, stream>>>(Wq, Wqp, Dd, Dd);
  pack_transpose_kernel<<<(Dd * Dd) / 256, 256, 0, stream>>>(Wk, Wkp, Dd, Dd);
  pack_transpose_kernel<<<(Dd * Dd) / 256, 256, 0, stream>>>(Wv, Wvp, Dd, Dd);
  pack_transpose_kernel<<<(Dd * Dd) / 256, 256, 0, stream>>>(Wo, Wop, Dd, Dd);
  pack_convert_kernel<<<(DFFf * Dd) / 256, 256, 0, stream>>>(W1, W1bf, DFFf * Dd);
  pack_convert_kernel<<<(Dd * DFFf) / 256, 256, 0, stream>>>(W2, W2bf, Dd * DFFf);

  // ---- QKV projections (f32 out for the VALU attention stages) ----
  gemm_bf16_kernel<0,1,0><<<dim3(Dd/128, ML/64), blk, 0, stream>>>(xbf, Wqp, bq,  nullptr, q,    nullptr, ML, Dd, Dd);
  gemm_bf16_kernel<0,1,0><<<dim3(Dd/128, ML/64), blk, 0, stream>>>(xbf, Wkp, bk,  nullptr, kbuf, nullptr, ML, Dd, Dd);
  gemm_bf16_kernel<0,1,0><<<dim3(Dd/128, ML/64), blk, 0, stream>>>(xbf, Wvp, bvp, nullptr, vbuf, nullptr, ML, Dd, Dd);

  // ---- ProbSparse selection ----
  sample_score_kernel<<<(Bb*Hh*Ls)/256, 256, 0, stream>>>(q, kbuf, idx, Msc);
  topk_kernel<<<Bb*Hh, 256, 0, stream>>>(Msc, mtop);
  vmean_kernel<<<(Bb*Hh*DVv + 255)/256, 256, 0, stream>>>(vbuf, vmean);
  attn_kernel<<<Bb*Hh*Uu, 256, 0, stream>>>(q, kbuf, vbuf, mtop, upd);
  ctx_bcast_kernel<<<(unsigned)(((size_t)Bb*Ls*Hh*DVv)/256), 256, 0, stream>>>(vmean, ctxbf);
  ctx_scatter_kernel<<<(Bb*Hh*Uu*DVv + 255)/256, 256, 0, stream>>>(mtop, upd, ctxbf);

  // ---- output projection (f32 + bf16 out) + FFN (GELU & residual fused) ----
  gemm_bf16_kernel<0,3,0><<<dim3(Dd/128,   ML/64), blk, 0, stream>>>(ctxbf,  Wop,  bo, nullptr, attn,    attnbf, ML, Dd,   Dd);
  gemm_bf16_kernel<1,2,0><<<dim3(DFFf/128, ML/64), blk, 0, stream>>>(attnbf, W1bf, b1, nullptr, nullptr, hbf,    ML, DFFf, Dd);
  gemm_bf16_kernel<0,1,1><<<dim3(Dd/128,   ML/64), blk, 0, stream>>>(hbf,    W2bf, b2, attn,    zbuf,    nullptr, ML, Dd,  DFFf);

  // ---- LayerNorm -> out, then m_top tail (twice) ----
  layernorm_kernel<<<ML, 256, 0, stream>>>(zbuf, gamma, beta, out);
  write_mtop_kernel<<<(Bb*Hh*Uu + 255)/256, 256, 0, stream>>>(mtop, out + (size_t)ML * Dd);

  (void)in_sizes; (void)n_in; (void)out_size; (void)ws_size;
}
